// KStore_28716151341458
// MI455X (gfx1250) — compile-verified, matrix operand order checked by reading, not run on hardware
//
#include <hip/hip_runtime.h>
#include <hip/hip_bf16.h>

typedef __attribute__((ext_vector_type(16))) __bf16 v16bf;
typedef __attribute__((ext_vector_type(8)))  __bf16 v8bf;
typedef __attribute__((ext_vector_type(8)))  float  v8f;

#define NQ       256
#define KDIM     512
#define KTOP     32
#define NSTRIPS  512          // 2 workgroups per WGP on a 256-WGP part; full util on larger parts
#define NEG_INF  (-1.0e30f)

// ---------------------------------------------------------------------------
// Phase 0: L2-normalize queries, store as bf16 (stays hot in 192MB L2).
// One block per query row; 256 threads x float2.
// ---------------------------------------------------------------------------
__global__ __launch_bounds__(256) void qnorm_kernel(const float* __restrict__ q,
                                                    __hip_bfloat16* __restrict__ qn) {
  __shared__ float red[256];
  const int tid = threadIdx.x, row = blockIdx.x;
  const float2* qr = (const float2*)(q + (size_t)row * KDIM);
  float2 v = qr[tid];
  red[tid] = v.x * v.x + v.y * v.y;
  __syncthreads();
  for (int off = 128; off > 0; off >>= 1) {
    if (tid < off) red[tid] += red[tid + off];
    __syncthreads();
  }
  const float rn = 1.0f / fmaxf(sqrtf(red[0]), 1e-12f);
  __hip_bfloat16* o = qn + (size_t)row * KDIM;
  o[2 * tid]     = __float2bfloat16(v.x * rn);
  o[2 * tid + 1] = __float2bfloat16(v.y * rn);
}

// ---------------------------------------------------------------------------
// Phase 1: bf16 WMMA similarity + per-strip running top-32.
// Block = 256 threads (8 waves). Each block owns a strip of key tiles and all
// 256 queries, so each key is streamed from HBM exactly once device-wide.
// Per iteration: stage 16 keys (f32->bf16) into LDS (prefetching the next
// tile), 8 waves x 2 query-tiles of 16x16 via v_wmma_f32_16x16x32_bf16 over
// K=512 (16 chained WMMAs; A fragments re-read from global (L2) each tile).
// ---------------------------------------------------------------------------
__global__ __launch_bounds__(256) void sim_topk_kernel(
    const float* __restrict__ keys, const __hip_bfloat16* __restrict__ qn,
    float* __restrict__ candS, int* __restrict__ candI,
    int ntilesTotal, int tilesPerStrip) {
  __shared__ __hip_bfloat16 sKeys[16 * KDIM];   // 16 KB: B tile, [key][K] bf16
  __shared__ float sScores[NQ * 16];            // 16 KB: [query][key_local]
  __shared__ float sTopS[NQ * KTOP];            // 32 KB
  __shared__ int   sTopI[NQ * KTOP];            // 32 KB

  const int tid  = threadIdx.x;
  const int strip = blockIdx.x;
  const int lane = tid & 31;
  const int wave = tid >> 5;
  const int m    = lane & 15;   // row (A) / key column (B) within tile
  const int hi   = lane >> 4;   // selects K sub-run per WMMA bf16 layout

  // init per-query top lists (thread tid owns query tid)
  for (int j = 0; j < KTOP; ++j) { sTopS[tid * KTOP + j] = NEG_INF; sTopI[tid * KTOP + j] = 0; }
  float curMin = NEG_INF;
  int   minPos = 0;
  __syncthreads();

  const int t0 = strip * tilesPerStrip;
  const int t1 = min(t0 + tilesPerStrip, ntilesTotal);

  for (int t = t0; t < t1; ++t) {
    // ---- stage 16 keys x 512 f32 -> bf16 into LDS (row-major [16][512]) ----
    const float4* src = (const float4*)(keys + (size_t)t * (16 * KDIM));
    for (int i = tid; i < (16 * KDIM) / 4; i += 256) {
      float4 f = src[i];
      __hip_bfloat16* d = sKeys + i * 4;
      d[0] = __float2bfloat16(f.x);
      d[1] = __float2bfloat16(f.y);
      d[2] = __float2bfloat16(f.z);
      d[3] = __float2bfloat16(f.w);
    }
    if (t + 1 < t1) {
      // prefetch next 32KB key tile (emits global_prefetch_b8)
      const char* nxt = (const char*)(keys + (size_t)(t + 1) * (16 * KDIM));
      __builtin_prefetch(nxt + tid * 128, 0, 1);
    }
    __syncthreads();

    // ---- WMMA: wave handles query tiles 2*wave and 2*wave+1 ----
    for (int sub = 0; sub < 2; ++sub) {
      const int qt = wave * 2 + sub;
      const __hip_bfloat16* arow = qn + (size_t)(qt * 16 + m) * KDIM;  // A: query row (global, L2-hot)
      const __hip_bfloat16* brow = sKeys + m * KDIM;                   // B: key column n==m (LDS)
      v8f c = {0.f, 0.f, 0.f, 0.f, 0.f, 0.f, 0.f, 0.f};
#pragma unroll
      for (int ks = 0; ks < KDIM / 32; ++ks) {
        const int kb = ks * 32 + hi * 8;  // 16-bit A/B layout: K runs [kb,kb+7] and [kb+16,kb+23]
        v8bf a0 = *(const v8bf*)(arow + kb);
        v8bf a1 = *(const v8bf*)(arow + kb + 16);
        v8bf b0 = *(const v8bf*)(brow + kb);
        v8bf b1 = *(const v8bf*)(brow + kb + 16);
        v16bf a = __builtin_shufflevector(a0, a1, 0,1,2,3,4,5,6,7,8,9,10,11,12,13,14,15);
        v16bf b = __builtin_shufflevector(b0, b1, 0,1,2,3,4,5,6,7,8,9,10,11,12,13,14,15);
        c = __builtin_amdgcn_wmma_f32_16x16x32_bf16(false, a, false, b, (short)0, c, false, false);
      }
      // C layout: VGPR r, lanes<16 -> M=r, lanes>=16 -> M=r+8; N=lane&15
#pragma unroll
      for (int r = 0; r < 8; ++r) {
        const int qidx = qt * 16 + r + hi * 8;
        sScores[qidx * 16 + m] = c[r];
      }
    }
    __syncthreads();

    // ---- per-query running top-32 (thread tid == query tid) ----
    const int nbase = t * 16;
    for (int j = 0; j < 16; ++j) {
      float s = sScores[tid * 16 + j];
      if (s > curMin) {
        sTopS[tid * KTOP + minPos] = s;
        sTopI[tid * KTOP + minPos] = nbase + j;
        float mn = sTopS[tid * KTOP];
        int   mp = 0;
        for (int u = 1; u < KTOP; ++u) {
          float vv = sTopS[tid * KTOP + u];
          if (vv < mn) { mn = vv; mp = u; }
        }
        curMin = mn; minPos = mp;
      }
    }
    __syncthreads();
  }

  // ---- emit candidates: [query][strip][KTOP] ----
  float* oS = candS + ((size_t)tid * NSTRIPS + strip) * KTOP;
  int*   oI = candI + ((size_t)tid * NSTRIPS + strip) * KTOP;
  for (int j = 0; j < KTOP; ++j) { oS[j] = sTopS[tid * KTOP + j]; oI[j] = sTopI[tid * KTOP + j]; }
}

// ---------------------------------------------------------------------------
// Phase 2: per-query merge of NSTRIPS x 32 candidates -> final top-32, then
// gather value rows. One block per query, 256 threads.
// ---------------------------------------------------------------------------
__global__ __launch_bounds__(256) void merge_gather_kernel(
    const float* __restrict__ candS, const int* __restrict__ candI,
    const float* __restrict__ values,
    float* __restrict__ outSlots, float* __restrict__ outScores,
    int* __restrict__ outIdx) {
  __shared__ float sS[NSTRIPS * KTOP];  // 64 KB
  __shared__ float rS[256];
  __shared__ int   rC[256];
  __shared__ float wS[KTOP];
  __shared__ int   wC[KTOP];
  __shared__ int   gIdx[KTOP];

  const int tid = threadIdx.x, q = blockIdx.x;
  const int NC = NSTRIPS * KTOP;

  for (int i = tid; i < NC; i += 256) sS[i] = candS[(size_t)q * NC + i];
  __syncthreads();

  for (int it = 0; it < KTOP; ++it) {
    float best = -2.0e30f;
    int   bc = 0;
    for (int i = tid; i < NC; i += 256) {
      float v = sS[i];
      if (v > best) { best = v; bc = i; }
    }
    rS[tid] = best; rC[tid] = bc;
    __syncthreads();
    for (int off = 128; off > 0; off >>= 1) {
      if (tid < off && rS[tid + off] > rS[tid]) { rS[tid] = rS[tid + off]; rC[tid] = rC[tid + off]; }
      __syncthreads();
    }
    if (tid == 0) { int c = rC[0]; wS[it] = rS[0]; wC[it] = c; sS[c] = -2.0e30f; }
    __syncthreads();
  }

  if (tid < KTOP) {
    const int c = wC[tid];
    const int idx = candI[(size_t)q * NC + c];
    gIdx[tid] = idx;
    outScores[q * KTOP + tid] = wS[tid];
    outIdx[q * KTOP + tid] = idx;  // int32 bit pattern, matches reference dtype
  }
  __syncthreads();

  for (int j = 0; j < KTOP; ++j) {
    const float4* src = (const float4*)(values + (size_t)gIdx[j] * KDIM);
    float4* dst = (float4*)(outSlots + ((size_t)q * KTOP + j) * KDIM);
    for (int e = tid; e < KDIM / 4; e += 256) dst[e] = src[e];
  }
}

// ---------------------------------------------------------------------------
extern "C" void kernel_launch(void* const* d_in, const int* in_sizes, int n_in,
                              void* d_out, int out_size, void* d_ws, size_t ws_size,
                              hipStream_t stream) {
  (void)n_in; (void)out_size; (void)ws_size;
  const float* query  = (const float*)d_in[0];
  const float* keys   = (const float*)d_in[1];
  const float* values = (const float*)d_in[2];
  // d_in[3] is k (== 32); fixed at compile time as KTOP.

  const long cap = (long)in_sizes[1] / KDIM;        // 500000
  const int ntiles = (int)(cap / 16);               // 31250
  const int tilesPerStrip = (ntiles + NSTRIPS - 1) / NSTRIPS;

  // workspace layout
  char* ws = (char*)d_ws;
  __hip_bfloat16* qn = (__hip_bfloat16*)ws;
  size_t off = (size_t)NQ * KDIM * sizeof(__hip_bfloat16);          // 256 KB
  float* candS = (float*)(ws + off);
  off += (size_t)NQ * NSTRIPS * KTOP * sizeof(float);               // 16 MB
  int* candI = (int*)(ws + off);                                    // 16 MB

  // output layout: slots | scores | indices
  float* outSlots  = (float*)d_out;
  float* outScores = outSlots + (size_t)NQ * KTOP * KDIM;
  int*   outIdx    = (int*)(outScores + (size_t)NQ * KTOP);

  qnorm_kernel<<<NQ, 256, 0, stream>>>(query, qn);
  sim_topk_kernel<<<NSTRIPS, 256, 0, stream>>>(keys, qn, candS, candI, ntiles, tilesPerStrip);
  merge_gather_kernel<<<NQ, 256, 0, stream>>>(candS, candI, values, outSlots, outScores, outIdx);
}
